// CalibrationLoss_60962765799469
// MI455X (gfx1250) — matrix operand order repaired and checked by reading, not compiled, not run on hardware
//
#include <hip/hip_runtime.h>
#include <hip/hip_fp16.h>

typedef __attribute__((ext_vector_type(16))) _Float16 v16h;
typedef __attribute__((ext_vector_type(8)))  float    v8f;

#define N_ROWS 32768
#define N_CLS  1000
#define NBINS  15
#define SHARP  20.0f
#define EPSF   1e-8f
#define LOG2E  1.4426950408889634f

// ---------------------------------------------------------------- zero accum
__global__ void calib_zero(float* __restrict__ acc) {
    int i = threadIdx.x;
    if (i < 48) acc[i] = 0.0f;
}

// ---------------------------------------------------------------- stage 1
// One wave per row. Branch-free 3-phase softmax-confidence:
//   1) lane max -> wave max M
//   2) s = sum exp2(x*log2e - M*log2e), 4 independent accumulators
//   3) argmax = min index where x == M (first occurrence)
// confidence = 1/s ; prediction correctness vs label.
__global__ __launch_bounds__(256) void calib_stage1(
        const float* __restrict__ logits,
        const int*   __restrict__ labels,
        float* __restrict__ conf_out,
        float* __restrict__ corr_out) {
    const int wave = threadIdx.x >> 5;
    const int lane = threadIdx.x & 31;
    const int row  = blockIdx.x * 8 + wave;
    const float* rp = logits + (size_t)row * N_CLS;

    const float NEG_INF = -__builtin_inff();

    // Load the lane's 32 values (1000 % 4 == 0, so vectors are all-or-nothing).
    float4 xs[8];
    #pragma unroll
    for (int it = 0; it < 8; ++it) {
        const int c = it * 128 + lane * 4;
        if (c < N_CLS) {
            xs[it] = *(const float4*)(rp + c);
        } else {
            xs[it] = make_float4(NEG_INF, NEG_INF, NEG_INF, NEG_INF);
        }
    }

    // Phase 1: lane max (tree), then wave max.
    float m = NEG_INF;
    #pragma unroll
    for (int it = 0; it < 8; ++it) {
        const float m01 = fmaxf(xs[it].x, xs[it].y);
        const float m23 = fmaxf(xs[it].z, xs[it].w);
        m = fmaxf(m, fmaxf(m01, m23));
    }
    #pragma unroll
    for (int off = 16; off > 0; off >>= 1) {
        m = fmaxf(m, __shfl_xor(m, off, 32));
    }
    // All lanes now hold the row max M.

    // Phase 2: sum of exps with 4 independent accumulators + first-occurrence argmax.
    const float negML = -m * LOG2E;
    float s0 = 0.0f, s1 = 0.0f, s2 = 0.0f, s3 = 0.0f;
    unsigned idx = 0xFFFFFFFFu;
    #pragma unroll
    for (int it = 0; it < 8; ++it) {
        const int c = it * 128 + lane * 4;
        const float x0 = xs[it].x, x1 = xs[it].y, x2 = xs[it].z, x3 = xs[it].w;
        s0 += exp2f(fmaf(x0, LOG2E, negML));
        s1 += exp2f(fmaf(x1, LOG2E, negML));
        s2 += exp2f(fmaf(x2, LOG2E, negML));
        s3 += exp2f(fmaf(x3, LOG2E, negML));
        idx = (x0 == m) ? min(idx, (unsigned)(c + 0)) : idx;
        idx = (x1 == m) ? min(idx, (unsigned)(c + 1)) : idx;
        idx = (x2 == m) ? min(idx, (unsigned)(c + 2)) : idx;
        idx = (x3 == m) ? min(idx, (unsigned)(c + 3)) : idx;
    }
    float s = (s0 + s1) + (s2 + s3);

    // Wave reduction: sum s, min idx.
    #pragma unroll
    for (int off = 16; off > 0; off >>= 1) {
        s   += __shfl_xor(s, off, 32);
        idx  = min(idx, (unsigned)__shfl_xor((int)idx, off, 32));
    }

    if (lane == 0) {
        conf_out[row] = 1.0f / s;                                 // max softmax prob
        corr_out[row] = ((int)idx == labels[row]) ? 1.0f : 0.0f;  // prediction correct
    }
}

// ---------------------------------------------------------------- stage 2
__device__ __forceinline__ float calib_sigmoid(float z) {
    // fast rcp is plenty for soft bin weights in [0,1]
    return __builtin_amdgcn_rcpf(1.0f + exp2f(-z * LOG2E));
}
__device__ __forceinline__ float calib_inbin(float c, int bin) {
    const float lo = (float)bin       * (1.0f / (float)NBINS);
    const float hi = (float)(bin + 1) * (1.0f / (float)NBINS);
    const float v  = calib_sigmoid(SHARP * (c - lo)) * calib_sigmoid(SHARP * (hi - c));
    return (bin < NBINS) ? v : 0.0f;                    // row 15 = padding
}

// D[16x16] += A(in_bin 16x32 f16) x B([correct|conf|ones|0...] 32x16 f16)
// accumulated over the 32768-row K dimension with v_wmma_f32_16x16x32_f16.
__global__ __launch_bounds__(256) void calib_stage2(
        const float* __restrict__ conf_g,
        const float* __restrict__ corr_g,
        float* __restrict__ accum) {
    const int lane   = threadIdx.x & 31;
    const int nWpb   = blockDim.x >> 5;
    const int waveId = blockIdx.x * nWpb + (threadIdx.x >> 5);
    const int nWaves = gridDim.x * nWpb;
    const int binN   = lane & 15;       // A: bin row M; B/D: column N
    const int hi     = lane >> 4;

    v8f c = {};
    const int nChunks = N_ROWS / 32;
    for (int ch = waveId; ch < nChunks; ch += nWaves) {
        const int r0 = ch * 32;
        const float myConf = conf_g[r0 + lane];
        const float myCorr = corr_g[r0 + lane];
        v16h a, b;
        #pragma unroll
        for (int v = 0; v < 8; ++v) {
            // A tile: 16-bit A 16x32 layout (VGPR v<4: K=2v(+hi*8); v>=4: K=16+2(v-4)(+hi*8))
            const int ka = ((v < 4) ? 2 * v : 16 + 2 * (v - 4)) + hi * 8;
            const float ca0 = __shfl(myConf, ka,     32);
            const float ca1 = __shfl(myConf, ka + 1, 32);
            a[2 * v]     = (_Float16)calib_inbin(ca0, binN);
            a[2 * v + 1] = (_Float16)calib_inbin(ca1, binN);
            // B tile: 32x16, lane = N, K same as A (v<4: 2v(+hi*8); v>=4: 16+2(v-4)(+hi*8)) + {0,1}
            const int kb = ((v < 4) ? 2 * v : 16 + 2 * (v - 4)) + hi * 8;
            const float bc0 = __shfl(myConf, kb,     32);
            const float bc1 = __shfl(myConf, kb + 1, 32);
            const float br0 = __shfl(myCorr, kb,     32);
            const float br1 = __shfl(myCorr, kb + 1, 32);
            const float v0 = (binN == 0) ? br0 : (binN == 1) ? bc0 : (binN == 2) ? 1.0f : 0.0f;
            const float v1 = (binN == 0) ? br1 : (binN == 1) ? bc1 : (binN == 2) ? 1.0f : 0.0f;
            b[2 * v]     = (_Float16)v0;
            b[2 * v + 1] = (_Float16)v1;
        }
        c = __builtin_amdgcn_wmma_f32_16x16x32_f16(
                /*neg_a=*/false, a, /*neg_b=*/false, b,
                /*c_mod=*/(short)0, c, /*reuse_a=*/false, /*reuse_b=*/false);
    }

    // D layout: VGPR j, lane -> M = j + 8*hi, N = lane&15
    #pragma unroll
    for (int j = 0; j < 8; ++j) {
        const int M = j + 8 * hi;
        if (M < NBINS && binN < 3) {
            atomicAdd(&accum[M * 3 + binN], c[j]);
        }
    }
}

// ---------------------------------------------------------------- finalize
__global__ void calib_finalize(const float* __restrict__ accum, float* __restrict__ out) {
    if (threadIdx.x == 0) {
        float loss = 0.0f;
        for (int k = 0; k < NBINS; ++k) {
            const float accN  = accum[k * 3 + 0];   // sum in_bin * correct
            const float confN = accum[k * 3 + 1];   // sum in_bin * conf
            const float mass  = accum[k * 3 + 2];   // sum in_bin
            const float d = (accN - confN) / (mass + EPSF);
            loss += d * d * mass;
        }
        out[0] = loss / (float)N_ROWS;
    }
}

// ---------------------------------------------------------------- launch
extern "C" void kernel_launch(void* const* d_in, const int* in_sizes, int n_in,
                              void* d_out, int out_size, void* d_ws, size_t ws_size,
                              hipStream_t stream) {
    const float* logits = (const float*)d_in[0];
    const int*   labels = (const int*)d_in[1];
    float* out = (float*)d_out;

    float* accum = (float*)d_ws;                               // 48 floats
    float* conf  = (float*)((char*)d_ws + 4096);               // 32768 floats
    float* corr  = conf + N_ROWS;                              // 32768 floats

    calib_zero    <<<1, 64, 0, stream>>>(accum);
    calib_stage1  <<<N_ROWS / 8, 256, 0, stream>>>(logits, labels, conf, corr);
    calib_stage2  <<<32, 256, 0, stream>>>(conf, corr, accum);
    calib_finalize<<<1, 32, 0, stream>>>(accum, out);
}
